// GroupedMoE_25005299598050
// MI455X (gfx1250) — matrix operand order repaired and checked by reading, not compiled, not run on hardware
//
#include <hip/hip_runtime.h>
#include <hip/hip_bf16.h>
#include <cstdint>

#define NTHREADS 256
#define BM 128
#define BN 64
#define BK 32
#define LDSW 20   // LDS row stride in dwords (80B: 16B-aligned, bank-spreading)

typedef __attribute__((ext_vector_type(16))) __bf16 v16bf;
typedef __attribute__((ext_vector_type(2)))  __bf16 bf16x2;
typedef __attribute__((ext_vector_type(8)))  float  v8f;

union Frag { v16bf v; unsigned u[8]; };

// Native f32->bf16 (RNE) conversions: lower to v_cvt_pk_bf16_f32.
__device__ __forceinline__ unsigned short f32_to_bf16_bits(float f) {
  union { __bf16 h; unsigned short s; } t;
  t.h = (__bf16)f;
  return t.s;
}
__device__ __forceinline__ unsigned pack_bf16x2(float lo, float hi) {
  union { bf16x2 h; unsigned u; } t;
  t.h[0] = (__bf16)lo;
  t.h[1] = (__bf16)hi;
  return t.u;
}

// Async DMA: copy 16 bytes global -> LDS (GLOBAL_LOAD_ASYNC_TO_LDS_B128,
// tracked by ASYNCcnt). LDS byte address = low 32 bits of generic pointer.
__device__ __forceinline__ void async_copy16(void* lds_dst, const void* gsrc) {
  unsigned lds_off = (unsigned)(uintptr_t)lds_dst;
  asm volatile("global_load_async_to_lds_b128 %0, %1, off"
               :
               : "v"(lds_off), "v"(gsrc)
               : "memory");
}
__device__ __forceinline__ void async_wait0() {
  asm volatile("s_wait_asynccnt 0x0" ::: "memory");
}

// Load 8 consecutive elements as 4 packed-bf16x2 dwords (converts if fp32).
template <typename T>
__device__ __forceinline__ uint4 load8_as_bf16(const T* p) {
  uint4 d;
  if constexpr (sizeof(T) == 2) {
    d = *(const uint4*)p;                       // global_load_b128
  } else {
    float4 f0 = *(const float4*)p;              // global_load_b128
    float4 f1 = *(const float4*)(p + 4);        // global_load_b128
    d.x = pack_bf16x2(f0.x, f0.y);
    d.y = pack_bf16x2(f0.z, f0.w);
    d.z = pack_bf16x2(f1.x, f1.y);
    d.w = pack_bf16x2(f1.z, f1.w);
  }
  return d;
}

// ---- fragment builders: exactly two ds_load_b128 each -----------------------
// A (16x32 bf16, ISA layout): lane=(m,half); u[v] = K-pair (v>>2)*8 + half*4 + (v&3)
__device__ __forceinline__ void build_a_frag(Frag& f, const unsigned (&lds)[BM][LDSW],
                                             int mrow, int hv) {
  uint4 q0 = *(const uint4*)&lds[mrow][hv * 4];
  uint4 q1 = *(const uint4*)&lds[mrow][8 + hv * 4];
  f.u[0] = q0.x; f.u[1] = q0.y; f.u[2] = q0.z; f.u[3] = q0.w;
  f.u[4] = q1.x; f.u[5] = q1.y; f.u[6] = q1.z; f.u[7] = q1.w;
}
// B (32x16 bf16, ISA layout): lane=(n,g); u[j] = K-pair (j>>2)*8 + g*4 + (j&3) of column n.
// LDS tile is stored column(n)-major: lds[n][kp] = packed K-pair kp of column n.
__device__ __forceinline__ void build_b_frag(Frag& f, const unsigned (&lds)[BN][LDSW],
                                             int ncol, int g) {
  uint4 q0 = *(const uint4*)&lds[ncol][g * 4];
  uint4 q1 = *(const uint4*)&lds[ncol][8 + g * 4];
  f.u[0] = q0.x; f.u[1] = q0.y; f.u[2] = q0.z; f.u[3] = q0.w;
  f.u[4] = q1.x; f.u[5] = q1.y; f.u[6] = q1.z; f.u[7] = q1.w;
}

// ---- tile stagers -----------------------------------------------------------
// A tile: 128x32, each thread handles 16 consecutive elements of one row.
// ASYNC (bf16 source only): DMA straight to LDS; else convert through VGPRs.
template <typename TA, bool GATHER, bool ASYNC>
__device__ __forceinline__ void stage_a(unsigned (&lds)[BM][LDSW], const TA* A,
                                        const int* ridx, int m0, int k0, int lda,
                                        int tid) {
  int r   = tid >> 1;             // row in tile
  int kp0 = (tid & 1) * 8;        // dword group (0 or 8)
  if constexpr (ASYNC && sizeof(TA) == 2 && !GATHER) {
    const TA* p = A + (size_t)(m0 + r) * lda + (k0 + kp0 * 2);
    async_copy16(&lds[r][kp0], p);
    async_copy16(&lds[r][kp0 + 4], p + 8);
    return;
  }
  uint4 d0, d1;
  bool ok = true;
  size_t row;
  if constexpr (GATHER) {
    int t = ridx[m0 + r];
    ok = (t >= 0);
    row = (size_t)(ok ? t : 0);
  } else {
    row = (size_t)(m0 + r);
  }
  if (ok) {
    const TA* p = A + row * lda + (k0 + kp0 * 2);
    d0 = load8_as_bf16(p);
    d1 = load8_as_bf16(p + 8);
  } else {
    d0 = make_uint4(0, 0, 0, 0);
    d1 = make_uint4(0, 0, 0, 0);
  }
  *(uint4*)&lds[r][kp0]     = d0;  // ds_store_b128
  *(uint4*)&lds[r][kp0 + 4] = d1;  // ds_store_b128
}

// B tile (K-major global: B[k*ldb+n]) -> LDS transposed to column-major pairs.
template <typename TB>
__device__ __forceinline__ void stage_b_kmajor(unsigned (&lds)[BN][LDSW], const TB* B,
                                               int n0, int k0, int ldb, int tid) {
  int k  = tid >> 3;              // 0..31
  int ng = (tid & 7) * 8;         // 8 consecutive columns
  uint4 d = load8_as_bf16(B + (size_t)(k0 + k) * ldb + (n0 + ng));
  unsigned vals[4] = {d.x, d.y, d.z, d.w};
#pragma unroll
  for (int j = 0; j < 4; ++j) {
    ((unsigned short*)&lds[ng + 2 * j][0])[k]     = (unsigned short)(vals[j] & 0xffffu);
    ((unsigned short*)&lds[ng + 2 * j + 1][0])[k] = (unsigned short)(vals[j] >> 16);
  }
}

// B tile (N-major global: B[n*ldb+k]) -> direct b128 store (already col-major).
template <typename TB>
__device__ __forceinline__ void stage_b_nmajor(unsigned (&lds)[BN][LDSW], const TB* B,
                                               int n0, int k0, int ldb, int tid) {
  int n   = tid >> 2;             // 0..63
  int kp0 = (tid & 3) * 4;        // dword group
  uint4 d = load8_as_bf16(B + (size_t)(n0 + n) * ldb + (k0 + kp0 * 2));
  *(uint4*)&lds[n][kp0] = d;      // ds_store_b128
}

// ---------------------------------------------------------------------------
// Dual-B WMMA GEMM with fused SwiGLU: Out = bf16(silu(A*B1) * (A*B2))
// ---------------------------------------------------------------------------
template <typename TA, typename TB, bool B_NMAJOR, bool GATHER>
__global__ __launch_bounds__(NTHREADS)
void wmma_gemm_dual_swiglu(const TA* __restrict__ Abase, long strideA,
                           const int* __restrict__ rowIdx, int rowStride,
                           const TB* __restrict__ B1base,
                           const TB* __restrict__ B2base, long strideB,
                           unsigned short* __restrict__ OutBase, long strideOut,
                           int Kdim, int lda, int ldb, int ldout) {
  const int e = blockIdx.z;
  const TA* A  = Abase  + (size_t)e * strideA;
  const TB* B1 = B1base + (size_t)e * strideB;
  const TB* B2 = B2base + (size_t)e * strideB;
  unsigned short* Out = OutBase + (size_t)e * strideOut;
  const int* ridx = GATHER ? (rowIdx + (size_t)e * rowStride) : nullptr;

  const int m0 = blockIdx.y * BM;
  const int n0 = blockIdx.x * BN;
  const int tid  = threadIdx.x;
  const int lane = tid & 31;
  const int wid  = tid >> 5;
  const int wm   = wid & 3;
  const int wn   = wid >> 2;

  __shared__ __align__(16) unsigned lds_a [BM][LDSW];
  __shared__ __align__(16) unsigned lds_b1[BN][LDSW];
  __shared__ __align__(16) unsigned lds_b2[BN][LDSW];

  v8f acc1[2][2], acc2[2][2];
#pragma unroll
  for (int i = 0; i < 2; ++i)
#pragma unroll
    for (int j = 0; j < 2; ++j)
#pragma unroll
      for (int r = 0; r < 8; ++r) { acc1[i][j][r] = 0.f; acc2[i][j][r] = 0.f; }

  for (int k0 = 0; k0 < Kdim; k0 += BK) {
    stage_a<TA, GATHER, false>(lds_a, A, ridx, m0, k0, lda, tid);
    if constexpr (B_NMAJOR) {
      stage_b_nmajor(lds_b1, B1, n0, k0, ldb, tid);
      stage_b_nmajor(lds_b2, B2, n0, k0, ldb, tid);
    } else {
      stage_b_kmajor(lds_b1, B1, n0, k0, ldb, tid);
      stage_b_kmajor(lds_b2, B2, n0, k0, ldb, tid);
    }
    if (k0 + BK < Kdim) {   // prefetch next K-tile of weights (global_prefetch_b8)
      if constexpr (B_NMAJOR) {
        __builtin_prefetch(B1 + (size_t)(n0 + (tid >> 2)) * ldb + (k0 + BK) + (tid & 3) * 8, 0, 3);
        __builtin_prefetch(B2 + (size_t)(n0 + (tid >> 2)) * ldb + (k0 + BK) + (tid & 3) * 8, 0, 3);
      } else {
        __builtin_prefetch(B1 + (size_t)(k0 + BK + (tid >> 3)) * ldb + n0 + (tid & 7) * 8, 0, 3);
        __builtin_prefetch(B2 + (size_t)(k0 + BK + (tid >> 3)) * ldb + n0 + (tid & 7) * 8, 0, 3);
      }
    }
    __syncthreads();

    Frag af[2], bfr1[2], bfr2[2];
    const int hv = lane >> 4;
#pragma unroll
    for (int fi = 0; fi < 2; ++fi)
      build_a_frag(af[fi], lds_a, wm * 32 + fi * 16 + (lane & 15), hv);
#pragma unroll
    for (int fj = 0; fj < 2; ++fj) {
      int ncol = wn * 32 + fj * 16 + (lane & 15);
      build_b_frag(bfr1[fj], lds_b1, ncol, hv);
      build_b_frag(bfr2[fj], lds_b2, ncol, hv);
    }
#pragma unroll
    for (int fi = 0; fi < 2; ++fi)
#pragma unroll
      for (int fj = 0; fj < 2; ++fj) {
        acc1[fi][fj] = __builtin_amdgcn_wmma_f32_16x16x32_bf16(
            false, af[fi].v, false, bfr1[fj].v, (short)0, acc1[fi][fj], false, false);
        acc2[fi][fj] = __builtin_amdgcn_wmma_f32_16x16x32_bf16(
            false, af[fi].v, false, bfr2[fj].v, (short)0, acc2[fi][fj], false, false);
      }
    __syncthreads();
  }

#pragma unroll
  for (int fi = 0; fi < 2; ++fi)
#pragma unroll
    for (int fj = 0; fj < 2; ++fj) {
      int n = n0 + wn * 32 + fj * 16 + (lane & 15);
#pragma unroll
      for (int r = 0; r < 8; ++r) {
        int m = m0 + wm * 32 + fi * 16 + ((lane >> 4) << 3) + r;
        float gv = acc1[fi][fj][r];
        float uv = acc2[fi][fj][r];
        float sv = gv / (1.0f + __expf(-gv));
        Out[(size_t)m * ldout + n] = f32_to_bf16_bits(sv * uv);
      }
    }
}

// ---------------------------------------------------------------------------
// Single-B WMMA GEMM (A is bf16: staged via async DMA to LDS).
// Epilogue: fp32 store, or gated atomic scatter-add.
// ---------------------------------------------------------------------------
template <typename TA, typename TB, bool B_NMAJOR, bool SCATTER>
__global__ __launch_bounds__(NTHREADS)
void wmma_gemm_single(const TA* __restrict__ Abase, long strideA,
                      const TB* __restrict__ Bbase, long strideB,
                      float* __restrict__ OutBase, long strideOut,
                      const int* __restrict__ rowIdx,
                      const float* __restrict__ rowGate, int rowStride,
                      int Kdim, int lda, int ldb, int ldout) {
  const int e = blockIdx.z;
  const TA* A = Abase + (size_t)e * strideA;
  const TB* B = Bbase + (size_t)e * strideB;
  float* Out  = OutBase + (size_t)e * strideOut;
  const int* ridx    = SCATTER ? (rowIdx + (size_t)e * rowStride) : nullptr;
  const float* rgate = SCATTER ? (rowGate + (size_t)e * rowStride) : nullptr;

  const int m0 = blockIdx.y * BM;
  const int n0 = blockIdx.x * BN;
  const int tid  = threadIdx.x;
  const int lane = tid & 31;
  const int wid  = tid >> 5;
  const int wm   = wid & 3;
  const int wn   = wid >> 2;

  __shared__ __align__(16) unsigned lds_a[BM][LDSW];
  __shared__ __align__(16) unsigned lds_b[BN][LDSW];

  v8f acc[2][2];
#pragma unroll
  for (int i = 0; i < 2; ++i)
#pragma unroll
    for (int j = 0; j < 2; ++j)
#pragma unroll
      for (int r = 0; r < 8; ++r) acc[i][j][r] = 0.f;

  for (int k0 = 0; k0 < Kdim; k0 += BK) {
    stage_a<TA, false, true>(lds_a, A, nullptr, m0, k0, lda, tid);
    if constexpr (B_NMAJOR) {
      stage_b_nmajor(lds_b, B, n0, k0, ldb, tid);
    } else {
      stage_b_kmajor(lds_b, B, n0, k0, ldb, tid);
    }
    if (k0 + BK < Kdim) {
      if constexpr (B_NMAJOR)
        __builtin_prefetch(B + (size_t)(n0 + (tid >> 2)) * ldb + (k0 + BK) + (tid & 3) * 8, 0, 3);
      else
        __builtin_prefetch(B + (size_t)(k0 + BK + (tid >> 3)) * ldb + n0 + (tid & 7) * 8, 0, 3);
    }
    if constexpr (sizeof(TA) == 2) async_wait0();  // A DMA done before barrier
    __syncthreads();

    Frag af[2], bfr[2];
    const int hv = lane >> 4;
#pragma unroll
    for (int fi = 0; fi < 2; ++fi)
      build_a_frag(af[fi], lds_a, wm * 32 + fi * 16 + (lane & 15), hv);
#pragma unroll
    for (int fj = 0; fj < 2; ++fj)
      build_b_frag(bfr[fj], lds_b, wn * 32 + fj * 16 + (lane & 15), hv);
#pragma unroll
    for (int fi = 0; fi < 2; ++fi)
#pragma unroll
      for (int fj = 0; fj < 2; ++fj)
        acc[fi][fj] = __builtin_amdgcn_wmma_f32_16x16x32_bf16(
            false, af[fi].v, false, bfr[fj].v, (short)0, acc[fi][fj], false, false);
    __syncthreads();
  }

#pragma unroll
  for (int fi = 0; fi < 2; ++fi)
#pragma unroll
    for (int fj = 0; fj < 2; ++fj) {
      int n = n0 + wn * 32 + fj * 16 + (lane & 15);
#pragma unroll
      for (int r = 0; r < 8; ++r) {
        int m = m0 + wm * 32 + fi * 16 + ((lane >> 4) << 3) + r;
        if constexpr (SCATTER) {
          int tok = ridx[m];
          if (tok >= 0)
            atomicAdd(&Out[(size_t)tok * ldout + n], acc[fi][fj][r] * rgate[m]);
        } else {
          Out[(size_t)m * ldout + n] = acc[fi][fj][r];
        }
      }
    }
}

// ---------------------------------------------------------------------------
// Router: one 64-thread block per token.
// ---------------------------------------------------------------------------
__global__ __launch_bounds__(64)
void router_kernel(const float* __restrict__ x, const float* __restrict__ rw,
                   float* __restrict__ gating, int* __restrict__ topk,
                   float* __restrict__ zacc) {
  const int t = blockIdx.x;
  __shared__ float sh[1024];
  __shared__ float slog[64];
  __shared__ float zsum;
  const int e = threadIdx.x;
  for (int i = e; i < 1024; i += 64) sh[i] = x[(size_t)t * 1024 + i];
  if (e == 0) zsum = 0.f;
  __syncthreads();
  const float* wr = rw + (size_t)e * 1024;
  float a = 0.f;
#pragma unroll 8
  for (int k = 0; k < 1024; ++k) a += sh[k] * wr[k];
  slog[e] = a;
  atomicAdd(&zsum, a * a);
  __syncthreads();
  if (e == 0) {
    atomicAdd(zacc, zsum);
    unsigned long long used = 0ull;
    float lg[6];
    int id[6];
    for (int k = 0; k < 6; ++k) {
      float best = -3.4e38f;
      int bi = 0;
      for (int j = 0; j < 64; ++j)
        if (((used >> j) & 1ull) == 0ull && slog[j] > best) { best = slog[j]; bi = j; }
      used |= (1ull << bi);
      id[k] = bi;
      lg[k] = best;
    }
    float mx = lg[0];
    for (int k = 1; k < 6; ++k) mx = fmaxf(mx, lg[k]);
    float ex[6], den = 0.f;
    for (int k = 0; k < 6; ++k) { ex[k] = __expf(lg[k] - mx); den += ex[k]; }
    float inv = 1.f / den;
    for (int k = 0; k < 6; ++k) {
      gating[(size_t)t * 6 + k] = ex[k] * inv;
      topk[(size_t)t * 6 + k]  = id[k];
    }
  }
}

__global__ void prep_kernel(float* __restrict__ routed, int* __restrict__ slot_token,
                            float* __restrict__ slot_gate, int* __restrict__ counts,
                            float* __restrict__ zacc, int nd, int ncap, int ne) {
  int i = blockIdx.x * blockDim.x + threadIdx.x;
  if (i < nd) routed[i] = 0.f;
  if (i < ncap) { slot_token[i] = -1; slot_gate[i] = 0.f; }
  if (i < ne) counts[i] = 0;
  if (i == 0) zacc[0] = 0.f;
}

__global__ void assign_kernel(const int* __restrict__ topk, const float* __restrict__ gating,
                              int* __restrict__ counts, int* __restrict__ slot_token,
                              float* __restrict__ slot_gate, int total) {
  int p = blockIdx.x * blockDim.x + threadIdx.x;
  if (p >= total) return;
  int e = topk[p];
  int slot = atomicAdd(&counts[e], 1);
  if (slot < 768) {
    slot_token[(size_t)e * 768 + slot] = p / 6;
    slot_gate[(size_t)e * 768 + slot]  = gating[p];
  }
}

__global__ void combine_kernel(const float* __restrict__ sh, const float* __restrict__ rt,
                               const float* __restrict__ zacc, float* __restrict__ out,
                               int total) {
  int i = blockIdx.x * blockDim.x + threadIdx.x;
  if (i < total) out[i] = (sh[i] + rt[i]) * 0.5f;
  if (i == 0) out[total] = zacc[0] * (1.0e-6f / (float)(4096 * 64));
}

extern "C" void kernel_launch(void* const* d_in, const int* in_sizes, int n_in,
                              void* d_out, int out_size, void* d_ws, size_t ws_size,
                              hipStream_t stream) {
  const float* x   = (const float*)d_in[0];  // [N, D]
  const float* rw  = (const float*)d_in[1];  // [E, D]
  const float* w12 = (const float*)d_in[2];  // [E, D, 2F]  K-major
  const float* w3  = (const float*)d_in[3];  // [E, F, D]   K-major
  const float* gw  = (const float*)d_in[4];  // [DFF, D]    N-major
  const float* uw  = (const float*)d_in[5];  // [DFF, D]    N-major
  const float* dw  = (const float*)d_in[6];  // [D, DFF]    N-major
  float* out = (float*)d_out;                // [N*D] + z_loss scalar
  (void)in_sizes; (void)n_in; (void)out_size; (void)ws_size;

  const int N = 4096, D = 1024, E = 64, K = 6, F = 1408, DFF = 4096, CAP = 768;

  char* ws = (char*)d_ws;
  size_t off = 0;
  auto alloc = [&](size_t bytes) -> void* {
    void* p = ws + off;
    off = (off + bytes + 255) & ~(size_t)255;
    return p;
  };
  unsigned short* act_s = (unsigned short*)alloc((size_t)N * DFF * 2);       // 32 MB
  unsigned short* act_e = (unsigned short*)alloc((size_t)E * CAP * F * 2);   // 138 MB
  float* sharedF    = (float*)alloc((size_t)N * D * 4);
  float* routed     = (float*)alloc((size_t)N * D * 4);
  float* gating     = (float*)alloc((size_t)N * K * 4);
  int*   topk       = (int*)alloc((size_t)N * K * 4);
  int*   slot_token = (int*)alloc((size_t)E * CAP * 4);
  float* slot_gate  = (float*)alloc((size_t)E * CAP * 4);
  int*   counts     = (int*)alloc((size_t)E * 4);
  float* zacc       = (float*)alloc(256);

  prep_kernel<<<(N * D + 255) / 256, 256, 0, stream>>>(routed, slot_token, slot_gate,
                                                       counts, zacc, N * D, E * CAP, E);
  router_kernel<<<N, 64, 0, stream>>>(x, rw, gating, topk, zacc);
  assign_kernel<<<(N * K + 255) / 256, 256, 0, stream>>>(topk, gating, counts,
                                                         slot_token, slot_gate, N * K);

  // Shared expert: act_s = silu(x @ gate^T) * (x @ up^T)   [N, DFF] bf16
  wmma_gemm_dual_swiglu<float, float, true, false>
      <<<dim3(DFF / BN, N / BM, 1), NTHREADS, 0, stream>>>(
          x, 0, nullptr, 0, gw, uw, 0, act_s, 0, D, D, D, DFF);

  // sharedF = act_s @ down^T   [N, D] fp32
  wmma_gemm_single<unsigned short, float, true, false>
      <<<dim3(D / BN, N / BM, 1), NTHREADS, 0, stream>>>(
          act_s, 0, dw, 0, sharedF, 0, nullptr, nullptr, 0, DFF, DFF, DFF, D);

  // Expert GEMM1 (gathered rows, per-expert slice of w12, fused SwiGLU)
  wmma_gemm_dual_swiglu<float, float, false, true>
      <<<dim3(F / BN, CAP / BM, E), NTHREADS, 0, stream>>>(
          x, 0, slot_token, CAP, w12, w12 + F, (long)D * 2 * F,
          act_e, (long)CAP * F, D, D, 2 * F, F);

  // Expert GEMM2 + gated atomic scatter-add into routed[token, :]
  wmma_gemm_single<unsigned short, float, false, true>
      <<<dim3(D / BN, CAP / BM, E), NTHREADS, 0, stream>>>(
          act_e, (long)CAP * F, w3, (long)F * D, routed, 0,
          slot_token, slot_gate, CAP, F, F, D, D);

  combine_kernel<<<(N * D + 255) / 256, 256, 0, stream>>>(sharedF, routed, zacc, out, N * D);
}
